// CLModel_6863357739229
// MI455X (gfx1250) — matrix operand order repaired and checked by reading, not compiled, not run on hardware
//
#include <hip/hip_runtime.h>
#include <stdint.h>

#define BB 32768
#define DD 1024
#define SS 4
#define CC 7
#define LL 256
#define DG 512

typedef __attribute__((ext_vector_type(16))) __bf16 v16bf;
typedef __attribute__((ext_vector_type(8)))  float  v8f;

union FragBF {
    v16bf v;
    uint4 q[2];
    unsigned short h[16];
};

__device__ __forceinline__ unsigned short f2bf(float f) {
    union { float f; unsigned u; } x; x.f = f;
    unsigned r = x.u + 0x7FFFu + ((x.u >> 16) & 1u);
    return (unsigned short)(r >> 16);
}
__device__ __forceinline__ float bf2f(unsigned short h) {
    union { unsigned u; float f; } x; x.u = ((unsigned)h) << 16;
    return x.f;
}
__device__ __forceinline__ v8f zero8() {
    v8f v;
    #pragma unroll
    for (int i = 0; i < 8; i++) v[i] = 0.0f;
    return v;
}

// ---------------------------------------------------------------- conversions
__global__ void k_cvt_bf16(const float* __restrict__ src,
                           unsigned short* __restrict__ dst, int n) {
    int i = blockIdx.x * blockDim.x + threadIdx.x;
    if (i < n) dst[i] = f2bf(src[i]);
}

// Write weight matrix [batch, K, Nsrc] (row-major f32) as bf16 in WMMA
// B-fragment layout: tiles of 32(K)x16(N); lane = ((k>>3)&1)*16 + (n&15);
// within lane, 16 K-values packed 2-per-dword (K even -> low half).
__global__ void k_swz(const float* __restrict__ src, unsigned short* __restrict__ dst,
                      int K, int Nsrc, int Npad, int total) {
    int i = blockIdx.x * blockDim.x + threadIdx.x;
    if (i >= total) return;
    int per = K * Npad;
    int s = i / per;
    int r = i - s * per;
    int k = r / Npad;
    int n = r - k * Npad;
    float v = (n < Nsrc) ? src[((size_t)s * K + k) * Nsrc + n] : 0.0f;
    int t    = (k >> 5) * (Npad >> 4) + (n >> 4);
    int lane = (((k >> 3) & 1) << 4) + (n & 15);
    int kin  = ((k >> 1) & 8) | (k & 7);
    size_t off = (size_t)s * per + ((size_t)(t * 32 + lane)) * 16 + (kin >> 1) * 2 + (kin & 1);
    dst[off] = f2bf(v);
}

// ---------------------------------------------------------------- anchors (tiny)
__global__ __launch_bounds__(256) void k_anchor(
    const float* __restrict__ emo,  // [C,S,D]
    const float* __restrict__ W1, const float* __restrict__ b1,
    const float* __restrict__ g1, const float* __restrict__ be1,
    const float* __restrict__ W2, const float* __restrict__ b2,
    float* __restrict__ ancN)       // [S,C,L] normalized
{
    const int c = blockIdx.x, s = blockIdx.y;
    const int tid = threadIdx.x;
    __shared__ float a_s[DD];
    __shared__ float h_s[DD];
    __shared__ float red[256];

    for (int i = tid; i < DD; i += 256) a_s[i] = emo[((size_t)c * SS + s) * DD + i];
    __syncthreads();

    float hv[4]; float lsum = 0.f, lsq = 0.f;
    for (int j = 0; j < 4; j++) {
        int e = tid + j * 256;
        const float* wp = W1 + (size_t)s * DD * DD + e;
        float acc = b1[s * DD + e];
        for (int d = 0; d < DD; d++) acc += a_s[d] * wp[(size_t)d * DD];
        hv[j] = acc; lsum += acc; lsq += acc * acc;
    }
    red[tid] = lsum; __syncthreads();
    for (int st = 128; st > 0; st >>= 1) { if (tid < st) red[tid] += red[tid + st]; __syncthreads(); }
    float mu = red[0] * (1.0f / DD); __syncthreads();
    red[tid] = lsq; __syncthreads();
    for (int st = 128; st > 0; st >>= 1) { if (tid < st) red[tid] += red[tid + st]; __syncthreads(); }
    float var = red[0] * (1.0f / DD) - mu * mu;
    float rs = rsqrtf(var + 1e-5f);
    __syncthreads();
    for (int j = 0; j < 4; j++) {
        int e = tid + j * 256;
        float v = (hv[j] - mu) * rs * g1[s * DD + e] + be1[s * DD + e];
        h_s[e] = fmaxf(v, 0.0f);
    }
    __syncthreads();
    const float* w2p = W2 + (size_t)s * DD * LL + tid;
    float acc = b2[s * LL + tid];
    for (int d = 0; d < DD; d++) acc += h_s[d] * w2p[(size_t)d * LL];
    red[tid] = acc * acc; __syncthreads();
    for (int st = 128; st > 0; st >>= 1) { if (tid < st) red[tid] += red[tid + st]; __syncthreads(); }
    float inv = 1.0f / fmaxf(sqrtf(red[0]), 1e-8f);
    ancN[((size_t)s * CC + c) * LL + tid] = acc * inv;
}

// ---------------------------------------------------------------- WMMA GEMM1 stage
// Block: 256 threads = 8 waves as 2(M)x4(N); wave tile 32x64 (2x4 WMMA tiles).
// Computes H[0:64, 0:NOUT] = Xb[m0:m0+64, 0:1024] * Wswz + bias, stages bf16 H to
// LDS (stride HSTR) and accumulates per-row sum / sumsq into rowsum/rowsumsq.
template <int NOUT, int HSTR>
__device__ __forceinline__ void gemm1_stage(
    const unsigned short* __restrict__ Xb,
    const unsigned short* __restrict__ Wswz,
    const float* __restrict__ bias,
    unsigned short* Hs, float* rowsum, float* rowsumsq,
    int m0, int tid)
{
    const int lane = tid & 31;
    const int wave = tid >> 5;
    const int wm = wave >> 2;     // 0..1
    const int wn = wave & 3;      // 0..3
    const int hi = lane >> 4;     // lane half
    const int ln = lane & 15;
    const int NT16 = NOUT >> 4;

    float ps[16], pss[16];
    #pragma unroll
    for (int i = 0; i < 16; i++) { ps[i] = 0.f; pss[i] = 0.f; }

    for (int np = 0; np < NOUT; np += 256) {
        v8f acc[2][4];
        #pragma unroll
        for (int mt = 0; mt < 2; mt++)
            #pragma unroll
            for (int nt = 0; nt < 4; nt++) acc[mt][nt] = zero8();

        for (int k0 = 0; k0 < 1024; k0 += 32) {
            FragBF a[2];
            #pragma unroll
            for (int mt = 0; mt < 2; mt++) {
                const unsigned short* p =
                    Xb + (size_t)(m0 + wm * 32 + mt * 16 + ln) * 1024 + k0 + hi * 8;
                a[mt].q[0] = *(const uint4*)(p);
                a[mt].q[1] = *(const uint4*)(p + 16);
            }
            FragBF b[4];
            #pragma unroll
            for (int nt = 0; nt < 4; nt++) {
                int n16 = (np >> 4) + wn * 4 + nt;
                const unsigned short* p =
                    Wswz + ((size_t)((k0 >> 5) * NT16 + n16) * 32 + lane) * 16;
                b[nt].q[0] = *(const uint4*)(p);
                b[nt].q[1] = *(const uint4*)(p + 8);
            }
            #pragma unroll
            for (int mt = 0; mt < 2; mt++)
                #pragma unroll
                for (int nt = 0; nt < 4; nt++)
                    acc[mt][nt] = __builtin_amdgcn_wmma_f32_16x16x32_bf16(
                        false, a[mt].v, false, b[nt].v, (short)0, acc[mt][nt],
                        false, false);
        }
        // epilogue: + bias, stage to LDS, accumulate row statistics
        #pragma unroll
        for (int nt = 0; nt < 4; nt++) {
            int n = np + wn * 64 + nt * 16 + ln;
            float bv = bias[n];
            #pragma unroll
            for (int mt = 0; mt < 2; mt++) {
                int mbase = wm * 32 + mt * 16 + hi * 8;
                #pragma unroll
                for (int r = 0; r < 8; r++) {
                    float h = acc[mt][nt][r] + bv;
                    Hs[(mbase + r) * HSTR + n] = f2bf(h);
                    ps[mt * 8 + r]  += h;
                    pss[mt * 8 + r] += h * h;
                }
            }
        }
    }
    // half-wave (16-lane) reductions, then LDS atomics
    #pragma unroll
    for (int i = 0; i < 16; i++) {
        float sv = ps[i], qv = pss[i];
        for (int m = 8; m >= 1; m >>= 1) {
            sv += __shfl_xor(sv, m, 32);
            qv += __shfl_xor(qv, m, 32);
        }
        if (ln == 0) {
            int mt = i >> 3, r = i & 7;
            int m = wm * 32 + mt * 16 + hi * 8 + r;
            atomicAdd(&rowsum[m], sv);
            atomicAdd(&rowsumsq[m], qv);
        }
    }
}

// ---------------------------------------------------------------- main fused kernel
__global__ __launch_bounds__(256) void k_main(
    const unsigned short* __restrict__ Xb,
    const unsigned short* __restrict__ W1b,
    const unsigned short* __restrict__ W2b,
    const float* __restrict__ b1g, const float* __restrict__ g1g,
    const float* __restrict__ be1g, const float* __restrict__ b2g,
    const float* __restrict__ ancN, float* __restrict__ probs)
{
    const int s  = blockIdx.y;
    const int m0 = blockIdx.x * 64;
    const int tid = threadIdx.x;

    __shared__ unsigned short Hs[64 * 1032];          // 129 KB, padded stride
    __shared__ float g1s[DD], be1s[DD];
    __shared__ float ancs[CC * LL];
    __shared__ float rowsum[64], rowsumsq[64], rowmu[64], rowrs[64];

    for (int i = tid; i < DD; i += 256) { g1s[i] = g1g[s * DD + i]; be1s[i] = be1g[s * DD + i]; }
    for (int i = tid; i < CC * LL; i += 256) ancs[i] = ancN[(size_t)s * CC * LL + i];
    if (tid < 64) { rowsum[tid] = 0.f; rowsumsq[tid] = 0.f; }
    __syncthreads();

    gemm1_stage<1024, 1032>(Xb, W1b + (size_t)s * DD * DD, b1g + s * DD,
                            Hs, rowsum, rowsumsq, m0, tid);
    __syncthreads();
    if (tid < 64) {
        float mu  = rowsum[tid] * (1.0f / DD);
        float var = rowsumsq[tid] * (1.0f / DD) - mu * mu;
        rowmu[tid] = mu;
        rowrs[tid] = rsqrtf(var + 1e-5f);
    }
    __syncthreads();

    // GEMM2: LN-ReLU(H) [64x1024] x W2 [1024x256]
    const int lane = tid & 31, wave = tid >> 5;
    const int wm = wave >> 2, wn = wave & 3;
    const int hi = lane >> 4, ln = lane & 15;
    v8f acc[2][4];
    #pragma unroll
    for (int mt = 0; mt < 2; mt++)
        #pragma unroll
        for (int nt = 0; nt < 4; nt++) acc[mt][nt] = zero8();

    const unsigned short* W2s = W2b + (size_t)s * DD * LL;
    for (int k0 = 0; k0 < 1024; k0 += 32) {
        int kb = k0 + hi * 8;
        FragBF a[2];
        #pragma unroll
        for (int mt = 0; mt < 2; mt++) {
            int mrow = wm * 32 + mt * 16 + ln;
            float mu = rowmu[mrow], rs = rowrs[mrow];
            const unsigned short* hp = &Hs[mrow * 1032 + kb];
            FragBF raw;
            raw.q[0] = *(const uint4*)(hp);
            raw.q[1] = *(const uint4*)(hp + 16);
            #pragma unroll
            for (int j = 0; j < 8; j++) {
                float v = bf2f(raw.h[j]);
                v = (v - mu) * rs * g1s[kb + j] + be1s[kb + j];
                a[mt].h[j] = f2bf(fmaxf(v, 0.0f));
            }
            #pragma unroll
            for (int j = 0; j < 8; j++) {
                float v = bf2f(raw.h[8 + j]);
                v = (v - mu) * rs * g1s[kb + 16 + j] + be1s[kb + 16 + j];
                a[mt].h[8 + j] = f2bf(fmaxf(v, 0.0f));
            }
        }
        FragBF b[4];
        #pragma unroll
        for (int nt = 0; nt < 4; nt++) {
            int n16 = wn * 4 + nt;
            const unsigned short* p =
                W2s + ((size_t)((k0 >> 5) * 16 + n16) * 32 + lane) * 16;
            b[nt].q[0] = *(const uint4*)(p);
            b[nt].q[1] = *(const uint4*)(p + 8);
        }
        #pragma unroll
        for (int mt = 0; mt < 2; mt++)
            #pragma unroll
            for (int nt = 0; nt < 4; nt++)
                acc[mt][nt] = __builtin_amdgcn_wmma_f32_16x16x32_bf16(
                    false, a[mt].v, false, b[nt].v, (short)0, acc[mt][nt],
                    false, false);
    }
    __syncthreads();                 // done reading Hs; reuse as f32 F buffer
    float* Fs = (float*)Hs;          // [64 x 260] padded
    #pragma unroll
    for (int nt = 0; nt < 4; nt++) {
        int n = wn * 64 + nt * 16 + ln;
        float bv = b2g[s * LL + n];
        #pragma unroll
        for (int mt = 0; mt < 2; mt++) {
            int mbase = wm * 32 + mt * 16 + hi * 8;
            #pragma unroll
            for (int r = 0; r < 8; r++)
                Fs[(mbase + r) * 260 + n] = acc[mt][nt][r] + bv;
        }
    }
    __syncthreads();
    // per-row: L2 norm, anchor dots, temp-softmax over classes
    if (tid < 64) {
        float ss = 0.f, dc[CC];
        #pragma unroll
        for (int c = 0; c < CC; c++) dc[c] = 0.f;
        for (int l = 0; l < LL; l++) {
            float v = Fs[tid * 260 + l];
            ss += v * v;
            #pragma unroll
            for (int c = 0; c < CC; c++) dc[c] += v * ancs[c * LL + l];
        }
        float inv = 1.0f / fmaxf(sqrtf(ss), 1e-8f);
        float sc[CC], mx = -1e30f;
        #pragma unroll
        for (int c = 0; c < CC; c++) {
            sc[c] = ((1.0f + dc[c] * inv) * 0.5f + 1e-8f) * 10.0f;  // /TEMP
            mx = fmaxf(mx, sc[c]);
        }
        float sum = 0.f, ev[CC];
        #pragma unroll
        for (int c = 0; c < CC; c++) { ev[c] = expf(sc[c] - mx); sum += ev[c]; }
        float isum = 1.0f / sum;
        int brow = m0 + tid;
        #pragma unroll
        for (int c = 0; c < CC; c++)
            probs[(size_t)brow * (SS * CC) + s * CC + c] = ev[c] * isum;
    }
}

// ---------------------------------------------------------------- gate + fusion
__global__ __launch_bounds__(256) void k_gate(
    const unsigned short* __restrict__ Xb,
    const unsigned short* __restrict__ Wg1b,
    const float* __restrict__ bg1, const float* __restrict__ ggg,
    const float* __restrict__ bgbg, const float* __restrict__ Wg2,
    const float* __restrict__ bg2, const float* __restrict__ probs,
    float* __restrict__ out)
{
    const int m0 = blockIdx.x * 64;
    const int tid = threadIdx.x;
    __shared__ unsigned short Hs[64 * 520];           // 65 KB
    __shared__ float ggs[DG], bgbs[DG];
    __shared__ float rowsum[64], rowsumsq[64], rowmu[64], rowrs[64];
    __shared__ float gates[64][4];

    for (int i = tid; i < DG; i += 256) { ggs[i] = ggg[i]; bgbs[i] = bgbg[i]; }
    if (tid < 64) { rowsum[tid] = 0.f; rowsumsq[tid] = 0.f; }
    __syncthreads();

    gemm1_stage<512, 520>(Xb, Wg1b, bg1, Hs, rowsum, rowsumsq, m0, tid);
    __syncthreads();
    if (tid < 64) {
        float mu  = rowsum[tid] * (1.0f / DG);
        float var = rowsumsq[tid] * (1.0f / DG) - mu * mu;
        rowmu[tid] = mu;
        rowrs[tid] = rsqrtf(var + 1e-5f);
    }
    __syncthreads();
    {   // tiny 512 -> 4 projection (VALU)
        int row = tid >> 2, j = tid & 3;
        float mu = rowmu[row], rs = rowrs[row];
        float acc = bg2[j];
        for (int k = 0; k < DG; k++) {
            float v = bf2f(Hs[row * 520 + k]);
            v = fmaxf((v - mu) * rs * ggs[k] + bgbs[k], 0.0f);
            acc += v * Wg2[k * 4 + j];
        }
        gates[row][j] = acc;
    }
    __syncthreads();
    if (tid < 64) {
        float g[SS], mx = -1e30f;
        #pragma unroll
        for (int s = 0; s < SS; s++) { g[s] = gates[tid][s]; mx = fmaxf(mx, g[s]); }
        float sum = 0.f;
        #pragma unroll
        for (int s = 0; s < SS; s++) { g[s] = expf(g[s] - mx); sum += g[s]; }
        float isum = 1.0f / sum;
        int brow = m0 + tid;
        float fused[CC];
        #pragma unroll
        for (int c = 0; c < CC; c++) fused[c] = 0.f;
        for (int s = 0; s < SS; s++) {
            float w = g[s] * isum;
            #pragma unroll
            for (int c = 0; c < CC; c++)
                fused[c] += w * probs[(size_t)brow * (SS * CC) + s * CC + c];
        }
        #pragma unroll
        for (int c = 0; c < CC; c++)
            out[(size_t)brow * CC + c] = logf(fused[c] + 1e-8f);
    }
}

// ---------------------------------------------------------------- launch
extern "C" void kernel_launch(void* const* d_in, const int* in_sizes, int n_in,
                              void* d_out, int out_size, void* d_ws, size_t ws_size,
                              hipStream_t stream) {
    (void)in_sizes; (void)n_in; (void)out_size; (void)ws_size;
    const float* mask = (const float*)d_in[0];
    const float* emo  = (const float*)d_in[1];
    const float* W1   = (const float*)d_in[2];
    const float* b1   = (const float*)d_in[3];
    const float* g1   = (const float*)d_in[4];
    const float* be1  = (const float*)d_in[5];
    const float* W2   = (const float*)d_in[6];
    const float* b2   = (const float*)d_in[7];
    const float* Wg1  = (const float*)d_in[8];
    const float* bg1  = (const float*)d_in[9];
    const float* gg   = (const float*)d_in[10];
    const float* bgb  = (const float*)d_in[11];
    const float* Wg2  = (const float*)d_in[12];
    const float* bg2  = (const float*)d_in[13];
    float* out = (float*)d_out;

    char* ws = (char*)d_ws;
    unsigned short* Xb   = (unsigned short*)ws;  ws += (size_t)BB * DD * 2;
    unsigned short* W1b  = (unsigned short*)ws;  ws += (size_t)SS * DD * DD * 2;
    unsigned short* W2b  = (unsigned short*)ws;  ws += (size_t)SS * DD * LL * 2;
    unsigned short* Wg1b = (unsigned short*)ws;  ws += (size_t)DD * DG * 2;
    float* ancN  = (float*)ws;                   ws += (size_t)SS * CC * LL * 4;
    float* probs = (float*)ws;                   ws += (size_t)BB * SS * CC * 4;

    const int nX = BB * DD;
    k_cvt_bf16<<<dim3((nX + 255) / 256), dim3(256), 0, stream>>>(mask, Xb, nX);

    const int nW1 = SS * DD * DD;
    k_swz<<<dim3((nW1 + 255) / 256), dim3(256), 0, stream>>>(W1, W1b, DD, DD, DD, nW1);
    const int nW2 = SS * DD * LL;
    k_swz<<<dim3((nW2 + 255) / 256), dim3(256), 0, stream>>>(W2, W2b, DD, LL, LL, nW2);
    const int nWg1 = DD * DG;
    k_swz<<<dim3((nWg1 + 255) / 256), dim3(256), 0, stream>>>(Wg1, Wg1b, DD, DG, DG, nWg1);

    k_anchor<<<dim3(CC, SS), dim3(256), 0, stream>>>(emo, W1, b1, g1, be1, W2, b2, ancN);

    k_main<<<dim3(BB / 64, SS), dim3(256), 0, stream>>>(Xb, W1b, W2b, b1, g1, be1, b2,
                                                        ancN, probs);

    k_gate<<<dim3(BB / 64), dim3(256), 0, stream>>>(Xb, Wg1b, bg1, gg, bgb, Wg2, bg2,
                                                    probs, out);
}
